// MultiGaussianKernel_67405216743453
// MI455X (gfx1250) — compile-verified, matrix operand order checked by reading, not run on hardware
//
#include <hip/hip_runtime.h>
#include <hip/hip_bf16.h>

// ---------------------------------------------------------------------------
// MultiGaussianKernel on MI455X (gfx1250):
//   d2[n,m] = max(0, x2[n] + y2[m] - 2*X.Y^T)   (Gram identity)
//   out     = sum_a exp(-d2/(2a)),  a in {1/8,1/4,1/2,1,2}
// GEMM via V_WMMA_F32_16X16X32_BF16 with a bf16 hi/lo split:
//   x = hi + lo (hi = truncated top 16 bits, lo = bf16(residual))
//   x.y ~= hi.hi + hi.lo + lo.hi   (3 WMMAs per fp32 K=32 chunk)
// ---------------------------------------------------------------------------

constexpr int D_DIM = 1024;   // feature dim (fixed by reference)
constexpr int BT    = 128;    // block tile (rows of X and rows of Y)
constexpr int KT    = 32;     // K chunk per stage (one bf16 WMMA depth)
constexpr int LDSS  = 40;     // LDS row stride in ushorts (32 data + 8 pad)

typedef __attribute__((ext_vector_type(16))) __bf16 v16bf;
typedef __attribute__((ext_vector_type(8)))  __bf16 v8bf;
typedef __attribute__((ext_vector_type(8)))  float  v8f;

// ---------------- row squared-norm kernel: one wave32 per row ----------------
__global__ __launch_bounds__(256) void rownorm_kernel(
    const float* __restrict__ X, const float* __restrict__ Y,
    float* __restrict__ x2, float* __restrict__ y2, int nx, int ny) {
  const int wave = threadIdx.x >> 5;
  const int lane = threadIdx.x & 31;
  const int row  = blockIdx.x * 8 + wave;
  const float* src;
  float* dst;
  if (row < nx) {
    src = X + (size_t)row * D_DIM; dst = x2 + row;
  } else {
    int r = row - nx;
    if (r >= ny) return;
    src = Y + (size_t)r * D_DIM; dst = y2 + r;
  }
  float s = 0.f;
#pragma unroll
  for (int j = 0; j < D_DIM / 128; ++j) {
    float4 v = *(const float4*)(src + j * 128 + lane * 4);
    s += v.x * v.x + v.y * v.y + v.z * v.z + v.w * v.w;
  }
#pragma unroll
  for (int off = 16; off > 0; off >>= 1) s += __shfl_xor(s, off, 32);
  if (lane == 0) *dst = s;
}

// ---------------- fused distance + multi-gaussian kernel ----------------
__global__ __launch_bounds__(256) void gauss_wmma_kernel(
    const float* __restrict__ X, const float* __restrict__ Y,
    const float* __restrict__ x2, const float* __restrict__ y2,
    float* __restrict__ out, int Mcols) {

  // hi/lo bf16 planes for the X tile (A) and Y tile (B)
  __shared__ unsigned short sXh[BT * LDSS];
  __shared__ unsigned short sXl[BT * LDSS];
  __shared__ unsigned short sYh[BT * LDSS];
  __shared__ unsigned short sYl[BT * LDSS];

  const int tid  = threadIdx.x;
  const int lane = tid & 31;
  const int wave = tid >> 5;
  const int brow = blockIdx.y * BT;   // X rows
  const int bcol = blockIdx.x * BT;   // Y rows (= output cols)

  // ----- loader mapping: thread t -> (row t/2, k-half 16*(t&1)) -----
  const int lr = tid >> 1;
  const int lc = (tid & 1) * 16;
  const float* gX = X + (size_t)(brow + lr) * D_DIM + lc;
  const float* gY = Y + (size_t)(bcol + lr) * D_DIM + lc;

  float fx[16], fy[16];

  auto loadg = [&](int kb) {
#pragma unroll
    for (int i = 0; i < 4; ++i) {
      float4 a = *(const float4*)(gX + kb + 4 * i);
      float4 b = *(const float4*)(gY + kb + 4 * i);
      fx[4 * i + 0] = a.x; fx[4 * i + 1] = a.y; fx[4 * i + 2] = a.z; fx[4 * i + 3] = a.w;
      fy[4 * i + 0] = b.x; fy[4 * i + 1] = b.y; fy[4 * i + 2] = b.z; fy[4 * i + 3] = b.w;
    }
  };

  // Split a pair of fp32 into packed bf16 hi (truncated) + bf16 lo (residual,
  // RNE via native fptrunc so the backend picks the hw cvt).
  // hi pack = one v_perm_b32 pulling the top 16 bits of both words.
  auto split_pair = [&](float f0, float f1, unsigned& hi, unsigned& lo) {
    unsigned u0 = __float_as_uint(f0);
    unsigned u1 = __float_as_uint(f1);
    hi = __builtin_amdgcn_perm(u1, u0, 0x07060302u);   // [u1.b3 u1.b2 u0.b3 u0.b2]
    float r0 = f0 - __uint_as_float(u0 & 0xFFFF0000u);
    float r1 = f1 - __uint_as_float(u1 & 0xFFFF0000u);
    union { __bf16 b[2]; unsigned u; } p;
    p.b[0] = (__bf16)r0;
    p.b[1] = (__bf16)r1;
    lo = p.u;
  };

  auto stash = [&]() {
    unsigned xh[8], xl[8], yh[8], yl[8];
#pragma unroll
    for (int i = 0; i < 8; ++i) {
      split_pair(fx[2 * i], fx[2 * i + 1], xh[i], xl[i]);
      split_pair(fy[2 * i], fy[2 * i + 1], yh[i], yl[i]);
    }
    const int base = lr * LDSS + lc;   // *2 bytes -> 16B aligned
    *(uint4*)&sXh[base]     = make_uint4(xh[0], xh[1], xh[2], xh[3]);
    *(uint4*)&sXh[base + 8] = make_uint4(xh[4], xh[5], xh[6], xh[7]);
    *(uint4*)&sXl[base]     = make_uint4(xl[0], xl[1], xl[2], xl[3]);
    *(uint4*)&sXl[base + 8] = make_uint4(xl[4], xl[5], xl[6], xl[7]);
    *(uint4*)&sYh[base]     = make_uint4(yh[0], yh[1], yh[2], yh[3]);
    *(uint4*)&sYh[base + 8] = make_uint4(yh[4], yh[5], yh[6], yh[7]);
    *(uint4*)&sYl[base]     = make_uint4(yl[0], yl[1], yl[2], yl[3]);
    *(uint4*)&sYl[base + 8] = make_uint4(yl[4], yl[5], yl[6], yl[7]);
  };

  // ----- compute mapping: 8 waves as 4 (row) x 2 (col); wave tile 32x64 -----
  const int wr    = (wave & 3) * 32;
  const int wc    = (wave >> 2) * 64;
  const int l16   = lane & 15;
  const int khalf = (lane >> 4) * 16;   // contiguous K: lane half h holds k 16h..16h+15 (A and B alike)

  v8f acc[2][4];
#pragma unroll
  for (int tr = 0; tr < 2; ++tr)
#pragma unroll
    for (int tc = 0; tc < 4; ++tc)
#pragma unroll
      for (int v = 0; v < 8; ++v) acc[tr][tc][v] = 0.f;

  auto frag = [&](const unsigned short* p) -> v16bf {
    union { v16bf v; v8bf h[2]; } u;
    u.h[0] = *(const v8bf*)(p);        // ds_load_b128
    u.h[1] = *(const v8bf*)(p + 8);    // ds_load_b128
    return u.v;
  };

  auto compute = [&]() {
    v16bf aH[2], aL[2];
#pragma unroll
    for (int tr = 0; tr < 2; ++tr) {
      const int r = wr + tr * 16 + l16;
      aH[tr] = frag(&sXh[r * LDSS + khalf]);
      aL[tr] = frag(&sXl[r * LDSS + khalf]);
    }
#pragma unroll
    for (int tc = 0; tc < 4; ++tc) {
      const int c = wc + tc * 16 + l16;
      v16bf bH = frag(&sYh[c * LDSS + khalf]);
      v16bf bL = frag(&sYl[c * LDSS + khalf]);
#pragma unroll
      for (int tr = 0; tr < 2; ++tr) {
        acc[tr][tc] = __builtin_amdgcn_wmma_f32_16x16x32_bf16(
            false, aH[tr], false, bH, (short)0, acc[tr][tc], false, false);
        acc[tr][tc] = __builtin_amdgcn_wmma_f32_16x16x32_bf16(
            false, aL[tr], false, bH, (short)0, acc[tr][tc], false, false);
        acc[tr][tc] = __builtin_amdgcn_wmma_f32_16x16x32_bf16(
            false, aH[tr], false, bL, (short)0, acc[tr][tc], false, false);
      }
    }
  };

  // ----- pipelined K loop: stash -> barrier -> (prefetch || wmma) -> barrier
  loadg(0);
  for (int k = 0; k < D_DIM / KT; ++k) {
    stash();
    __syncthreads();
    if (k + 1 < D_DIM / KT) loadg((k + 1) * KT);
    compute();
    __syncthreads();
  }

  // ----- fused epilogue: d2 -> sum of 5 gaussians -> NT store -----
  float xr[2][8];
#pragma unroll
  for (int tr = 0; tr < 2; ++tr)
#pragma unroll
    for (int v = 0; v < 8; ++v)
      xr[tr][v] = x2[brow + wr + tr * 16 + (lane >> 4) * 8 + v];

#pragma unroll
  for (int tc = 0; tc < 4; ++tc) {
    const int col = bcol + wc + tc * 16 + l16;
    const float yv = y2[col];
#pragma unroll
    for (int tr = 0; tr < 2; ++tr) {
      const int rbase = brow + wr + tr * 16 + (lane >> 4) * 8;
#pragma unroll
      for (int v = 0; v < 8; ++v) {
        float d2 = xr[tr][v] + yv - 2.0f * acc[tr][tc][v];
        d2 = fmaxf(d2, 0.0f);
        float s = __expf(-4.0f  * d2)    // alpha = 1/8
                + __expf(-2.0f  * d2)    // alpha = 1/4
                + __expf(-1.0f  * d2)    // alpha = 1/2
                + __expf(-0.5f  * d2)    // alpha = 1
                + __expf(-0.25f * d2);   // alpha = 2
        // 64 MB streamed output, never re-read: keep it out of L2 so the
        // X/Y tiles (32 MB) stay resident across all 1024 blocks.
        __builtin_nontemporal_store(s, &out[(size_t)(rbase + v) * Mcols + col]);
      }
    }
  }
}

extern "C" void kernel_launch(void* const* d_in, const int* in_sizes, int n_in,
                              void* d_out, int out_size, void* d_ws, size_t ws_size,
                              hipStream_t stream) {
  const float* X = (const float*)d_in[0];
  const float* Y = (const float*)d_in[1];
  float* out = (float*)d_out;

  const int N = in_sizes[0] / D_DIM;   // 4096 rows of X
  const int M = in_sizes[1] / D_DIM;   // 4096 rows of Y

  float* x2 = (float*)d_ws;            // N floats
  float* y2 = x2 + N;                  // M floats (32 KB total workspace)

  const int totalRows = N + M;
  rownorm_kernel<<<(totalRows + 7) / 8, 256, 0, stream>>>(X, Y, x2, y2, N, M);

  dim3 grid(M / BT, N / BT);           // 32 x 32 blocks of 128x128
  gauss_wmma_kernel<<<grid, 256, 0, stream>>>(X, Y, x2, y2, out, M);
}
